// TriplaneAttention_72962904424786
// MI455X (gfx1250) — matrix operand order repaired and checked by reading, not compiled, not run on hardware
//
#include <hip/hip_runtime.h>
#include <hip/hip_bf16.h>
#include <cstdint>

// ---------------------------------------------------------------------------
// TriplaneAttention for MI455X (gfx1250, wave32, WMMA bf16 16x16x32).
// Mask structure: each query (p,i,j) attends exactly 64 keys = two 32-key
// arithmetic sequences (group A fixed by i, group B fixed by j), so all
// attention math is dense 32x32x64 / 32x64x32 WMMA GEMMs.
// LDS staging uses CDNA5 async DMA (global_load_async_to_lds_b128,
// ASYNCcnt-tracked), double-buffered in the big GEMM so the DMA engine
// runs concurrently with the WMMA pipe.
// ---------------------------------------------------------------------------

typedef __bf16 bf16;
typedef __attribute__((ext_vector_type(16))) __bf16 bf16x16;
typedef __attribute__((ext_vector_type(8)))  float  f32x8;

#define B_  2
#define RES 32
#define NQ  3072      // 3*32*32
#define C_  512
#define H_  8
#define D_  64
#define BN  (B_ * NQ) // 6144 rows for all big GEMMs

// ----------------------------- CDNA5 helpers -------------------------------

// Async DMA 16B from global to LDS (per lane). Generic shared pointers carry
// the LDS byte offset in addr[31:0], which is what the VDST operand wants.
__device__ __forceinline__ void async_copy_b128(const void* gsrc, void* ldst) {
  unsigned loff = (unsigned)(uintptr_t)ldst;
  asm volatile("global_load_async_to_lds_b128 %0, %1, off"
               :: "v"(loff), "v"(gsrc) : "memory");
}
__device__ __forceinline__ void wait_async() {
  asm volatile("s_wait_asynccnt 0" ::: "memory");
}

__device__ __forceinline__ f32x8 wmma_bf16(bf16x16 a, bf16x16 b, f32x8 c) {
  // 8 args: (neg_a, A, neg_b, B, c_mod, C, reuse_a, reuse_b)
  return __builtin_amdgcn_wmma_f32_16x16x32_bf16(false, a, false, b,
                                                 (short)0, c, false, false);
}

// A-matrix fragment (16x32, MxK) from LDS stored row-major [m][k], ld elems.
// Layout: lanes 0-15 -> M=0..15 with K = 0..15;
//         lanes 16-31 -> same M with K = 16..31.
__device__ __forceinline__ bf16x16 load_frag_a(const bf16* smem, int m0,
                                               int k0, int ld) {
  int lane = threadIdx.x & 31;
  const bf16* p = smem + (m0 + (lane & 15)) * ld + k0 + ((lane >> 4) << 4);
  union { uint4 q[2]; bf16x16 v; } u;
  u.q[0] = *reinterpret_cast<const uint4*>(p);       // K = kh+0..7
  u.q[1] = *reinterpret_cast<const uint4*>(p + 8);   // K = kh+8..15
  return u.v;
}

// B-matrix fragment (32x16, KxN) when LDS holds it TRANSPOSED, i.e. [n][k]
// row-major (natural layout for W[o][k] and for K-vectors).
__device__ __forceinline__ bf16x16 load_frag_b(const bf16* smem, int n0,
                                               int k0, int ld) {
  int lane = threadIdx.x & 31;
  const bf16* p = smem + (n0 + (lane & 15)) * ld + k0 + ((lane >> 4) << 4);
  union { uint4 q[2]; bf16x16 v; } u;
  u.q[0] = *reinterpret_cast<const uint4*>(p);
  u.q[1] = *reinterpret_cast<const uint4*>(p + 8);
  return u.v;
}

// B-matrix fragment when LDS holds it in natural [k][n] order (V tiles).
__device__ __forceinline__ bf16x16 load_frag_b_kn(const bf16* smem, int n0,
                                                  int k0, int ld) {
  int lane = threadIdx.x & 31;
  int n  = n0 + (lane & 15);
  int kh = k0 + ((lane >> 4) << 4);
  bf16x16 f;
#pragma unroll
  for (int e = 0; e < 16; ++e) f[e] = smem[(kh + e) * ld + n];
  return f;
}

// --------------------------- elementwise kernels ---------------------------

__global__ void f32_to_bf16_kernel(const float4* __restrict__ src,
                                   uint2* __restrict__ dst, int n4) {
  int i = blockIdx.x * blockDim.x + threadIdx.x;
  if (i >= n4) return;
  float4 v = src[i];
  bf16 o[4] = {(bf16)v.x, (bf16)v.y, (bf16)v.z, (bf16)v.w};
  dst[i] = *reinterpret_cast<uint2*>(o);
}

__global__ void add_to_bf16_kernel(const float4* __restrict__ a,
                                   const float4* __restrict__ b,
                                   uint2* __restrict__ o, int n4) {
  int i = blockIdx.x * blockDim.x + threadIdx.x;
  if (i >= n4) return;
  float4 x = a[i], y = b[i];
  bf16 r[4] = {(bf16)(x.x + y.x), (bf16)(x.y + y.y),
               (bf16)(x.z + y.z), (bf16)(x.w + y.w)};
  o[i] = *reinterpret_cast<uint2*>(r);
}

// ------------------------ big GEMM: out = A @ W^T ---------------------------
// A: [BN x 512] bf16, W: [512 x 512] bf16 row-major (o,k).
// out[m,o] = scale * sum_k A[m,k] W[o,k] (+ bias[o]).
// Block tile 128x128, BK=32, 8 waves (4M x 2N), wave tile 32x64.
// Double-buffered async-DMA pipeline: while tile i is multiplied, tile i+1
// streams into the other LDS buffer via the async engine.

template <bool OUT_BF16, bool HAS_BIAS>
__global__ __launch_bounds__(256) void wmma_gemm_kernel(
    const bf16* __restrict__ A, const bf16* __restrict__ W,
    const float* __restrict__ bias, void* __restrict__ out, float scale) {
  constexpr int K = 512, Nd = 512, LD = 40;  // pad 8 keeps 16B alignment
  constexpr int KSTEPS = K / 32;
  __shared__ bf16 As[2][128 * LD];
  __shared__ bf16 Bs[2][128 * LD];
  const int m0 = blockIdx.x * 128;
  const int n0 = blockIdx.y * 128;
  const int t  = threadIdx.x;
  const int w  = t >> 5;
  const int wm = (w & 3) * 32;
  const int wn = (w >> 2) * 64;

  // Each thread DMAs 2 x 16B per tile (4096 bf16 = 256 threads * 16 elems).
  auto stage = [&](int k0, int buf) {
#pragma unroll
    for (int it = 0; it < 2; ++it) {
      int idx = (it * 256 + t) * 8;
      int row = idx >> 5, col = idx & 31;
      async_copy_b128(&A[(size_t)(m0 + row) * K + k0 + col],
                      &As[buf][row * LD + col]);
      async_copy_b128(&W[(size_t)(n0 + row) * K + k0 + col],
                      &Bs[buf][row * LD + col]);
    }
  };

  f32x8 acc[2][4] = {};
  stage(0, 0);

  for (int i = 0; i < KSTEPS; ++i) {
    const int p = i & 1;
    wait_async();      // tile i resident in buffer p
    __syncthreads();   // all waves past reads of buffer p^1 (prev iter)
    if (i + 1 < KSTEPS) stage((i + 1) * 32, p ^ 1);

    bf16x16 bfrag[4];
#pragma unroll
    for (int nt = 0; nt < 4; ++nt)
      bfrag[nt] = load_frag_b(Bs[p], wn + nt * 16, 0, LD);
#pragma unroll
    for (int mt = 0; mt < 2; ++mt) {
      bf16x16 afrag = load_frag_a(As[p], wm + mt * 16, 0, LD);
#pragma unroll
      for (int nt = 0; nt < 4; ++nt)
        acc[mt][nt] = wmma_bf16(afrag, bfrag[nt], acc[mt][nt]);
    }
  }

  const int lane = t & 31;
  const int cl = lane & 15, rh = (lane >> 4) * 8;
  float bcol[4];
  if (HAS_BIAS) {
#pragma unroll
    for (int nt = 0; nt < 4; ++nt) bcol[nt] = bias[n0 + wn + nt * 16 + cl];
  }
#pragma unroll
  for (int mt = 0; mt < 2; ++mt)
#pragma unroll
    for (int nt = 0; nt < 4; ++nt)
#pragma unroll
      for (int v = 0; v < 8; ++v) {
        int row = wm + mt * 16 + rh + v;
        int col = wn + nt * 16 + cl;
        float val = acc[mt][nt][v] * scale + (HAS_BIAS ? bcol[nt] : 0.0f);
        size_t o = (size_t)(m0 + row) * Nd + n0 + col;
        if (OUT_BF16) ((bf16*)out)[o] = (bf16)val;
        else          ((float*)out)[o] = val;
      }
}

// ------------------------- triplane key geometry ---------------------------
// Query (p,i,j): group A keys (pass 0) depend only on i, group B (pass 1)
// only on j; keys are n = kb + c*ks, c = 0..31.
// Queries of a tile: n = nbase + row*rstep with rstep in {1, 32}.

__device__ __forceinline__ void key_base(int p, int r, int pass,
                                         int& kb, int& ks) {
  if (pass == 0) {
    if      (p == 0) { kb = 1024 + r * 32; ks = 1;  }
    else if (p == 1) { kb = r * 32;        ks = 1;  }
    else             { kb = r;             ks = 32; }
  } else {
    if      (p == 0) { kb = 2048 + r * 32; ks = 1;  }
    else if (p == 1) { kb = 2048 + r;      ks = 32; }
    else             { kb = 1024 + r;      ks = 32; }
  }
}

// ------------------- scores: S[q, 0..63] = scale*(q . k) -------------------
// One wave per (b,h,p,r,pass): 32x32x64 GEMM. Q already carries D^-1/2.

__global__ __launch_bounds__(32) void attn_score_kernel(
    const bf16* __restrict__ Q, const bf16* __restrict__ Kv,
    float* __restrict__ S) {
  const int x = blockIdx.x, p = x >> 5, r = x & 31;
  const int h = blockIdx.y;
  const int b = blockIdx.z >> 1, pass = blockIdx.z & 1;
  constexpr int LDQ = 72;
  __shared__ bf16 Qs[32 * LDQ];
  __shared__ bf16 Ks[32 * LDQ];
  const int t = threadIdx.x;

  const int nbase = p * 1024 + (pass ? r : r * 32);
  const int rstep = pass ? 32 : 1;
  int kb, ks; key_base(p, r, pass, kb, ks);
  {
    const bf16* qg = Q + ((size_t)(b * NQ + nbase + t * rstep)) * C_ + h * D_;
    const bf16* kg = Kv + ((size_t)(b * NQ + kb + t * ks)) * C_ + h * D_;
#pragma unroll
    for (int e = 0; e < 64; e += 8) {
      async_copy_b128(qg + e, &Qs[t * LDQ + e]);
      async_copy_b128(kg + e, &Ks[t * LDQ + e]);
    }
  }
  wait_async();
  __syncthreads();

  f32x8 acc[2][2] = {};
#pragma unroll
  for (int k0 = 0; k0 < 64; k0 += 32)
#pragma unroll
    for (int mt = 0; mt < 2; ++mt) {
      bf16x16 afrag = load_frag_a(Qs, mt * 16, k0, LDQ);
#pragma unroll
      for (int nt = 0; nt < 2; ++nt) {
        bf16x16 bfrag = load_frag_b(Ks, nt * 16, k0, LDQ);  // keys are [n][k]
        acc[mt][nt] = wmma_bf16(afrag, bfrag, acc[mt][nt]);
      }
    }

  const int cl = t & 15, rh = (t >> 4) * 8;
  // S row for query row 'row' is affine: base + row * rstep * 64 floats.
  float* srow = S + ((size_t)(b * H_ + h) * NQ + nbase) * 64 + pass * 32;
  const int rpitch = rstep * 64;
#pragma unroll
  for (int mt = 0; mt < 2; ++mt)
#pragma unroll
    for (int nt = 0; nt < 2; ++nt)
#pragma unroll
      for (int v = 0; v < 8; ++v) {
        int row = mt * 16 + rh + v;
        srow[(size_t)row * rpitch + nt * 16 + cl] = acc[mt][nt][v];
      }
}

// ------------------------- softmax over the 64 keys ------------------------

__global__ void softmax64_kernel(const float* __restrict__ S,
                                 bf16* __restrict__ P, int total) {
  int q = blockIdx.x * blockDim.x + threadIdx.x;
  if (q >= total) return;
  const float* s = S + (size_t)q * 64;
  float mx = s[0];
#pragma unroll
  for (int i = 1; i < 64; ++i) mx = fmaxf(mx, s[i]);
  float e[64], sum = 0.0f;
#pragma unroll
  for (int i = 0; i < 64; ++i) { e[i] = __expf(s[i] - mx); sum += e[i]; }
  float inv = 1.0f / sum;
  bf16* o = P + (size_t)q * 64;
#pragma unroll
  for (int i = 0; i < 64; ++i) o[i] = (bf16)(e[i] * inv);
}

// --------------------- PV: O += P(32x32) @ Vkeys(32x64) --------------------

__global__ __launch_bounds__(32) void attn_pv_kernel(
    const bf16* __restrict__ P, const bf16* __restrict__ V,
    float* __restrict__ Oa, float* __restrict__ Ob) {
  const int x = blockIdx.x, p = x >> 5, r = x & 31;
  const int h = blockIdx.y;
  const int b = blockIdx.z >> 1, pass = blockIdx.z & 1;
  constexpr int LDP = 40, LDV = 72;
  __shared__ bf16 Ps[32 * LDP];
  __shared__ bf16 Vs[32 * LDV];
  const int t = threadIdx.x;

  const int nbase = p * 1024 + (pass ? r : r * 32);
  const int rstep = pass ? 32 : 1;
  int kb, ks; key_base(p, r, pass, kb, ks);
  {
    const bf16* pg =
        P + ((size_t)((b * H_ + h) * NQ + nbase + t * rstep)) * 64 + pass * 32;
#pragma unroll
    for (int e = 0; e < 32; e += 8)
      async_copy_b128(pg + e, &Ps[t * LDP + e]);
    const bf16* vg = V + ((size_t)(b * NQ + kb + t * ks)) * C_ + h * D_;
#pragma unroll
    for (int e = 0; e < 64; e += 8)
      async_copy_b128(vg + e, &Vs[t * LDV + e]);
  }
  wait_async();
  __syncthreads();

  f32x8 acc[2][4] = {};
#pragma unroll
  for (int mt = 0; mt < 2; ++mt) {
    bf16x16 afrag = load_frag_a(Ps, mt * 16, 0, LDP);
#pragma unroll
    for (int nt = 0; nt < 4; ++nt) {
      bf16x16 bfrag = load_frag_b_kn(Vs, nt * 16, 0, LDV);  // V is [k=c][n=d]
      acc[mt][nt] = wmma_bf16(afrag, bfrag, acc[mt][nt]);
    }
  }

  float* orow = (pass ? Ob : Oa) + ((size_t)(b * NQ + nbase)) * C_ + h * D_;
  const size_t rpitch = (size_t)rstep * C_;
  const int cl = t & 15, rh = (t >> 4) * 8;
#pragma unroll
  for (int mt = 0; mt < 2; ++mt)
#pragma unroll
    for (int nt = 0; nt < 4; ++nt)
#pragma unroll
      for (int v = 0; v < 8; ++v) {
        int row = mt * 16 + rh + v;
        orow[row * rpitch + nt * 16 + cl] = acc[mt][nt][v];
      }
}

// ------------------------------- launcher ----------------------------------

static inline size_t align256(size_t x) { return (x + 255) & ~(size_t)255; }

extern "C" void kernel_launch(void* const* d_in, const int* in_sizes, int n_in,
                              void* d_out, int out_size, void* d_ws, size_t ws_size,
                              hipStream_t stream) {
  (void)in_sizes; (void)n_in; (void)out_size; (void)ws_size;
  const float* x  = (const float*)d_in[0];
  const float* Wq = (const float*)d_in[1];
  const float* Wk = (const float*)d_in[2];
  const float* Wv = (const float*)d_in[3];
  const float* Wp = (const float*)d_in[4];
  const float* bp = (const float*)d_in[5];
  float* out = (float*)d_out;

  char* ws = (char*)d_ws;
  size_t off = 0;
  auto take = [&](size_t bytes) { char* p = ws + off; off = align256(off + bytes); return p; };

  const size_t BNC = (size_t)BN * C_;       // 6144*512
  const size_t CC  = (size_t)C_ * C_;
  const size_t NQS = (size_t)B_ * H_ * NQ;  // 49152 queries

  bf16* xh  = (bf16*)take(BNC * 2);
  bf16* wqh = (bf16*)take(CC * 2);
  bf16* wkh = (bf16*)take(CC * 2);
  bf16* wvh = (bf16*)take(CC * 2);
  bf16* wph = (bf16*)take(CC * 2);
  bf16* qh  = (bf16*)take(BNC * 2);
  bf16* kh  = (bf16*)take(BNC * 2);
  bf16* vh  = (bf16*)take(BNC * 2);
  float* S  = (float*)take(NQS * 64 * 4);
  bf16* P   = (bf16*)take(NQS * 64 * 2);
  float* Oa = (float*)take(BNC * 4);
  float* Ob = (float*)take(BNC * 4);
  bf16* Oh  = (bf16*)take(BNC * 2);

  // 1) fp32 -> bf16 (vectorized x4)
  f32_to_bf16_kernel<<<(int)(BNC / 4 / 256), 256, 0, stream>>>(
      (const float4*)x, (uint2*)xh, (int)(BNC / 4));
  f32_to_bf16_kernel<<<(int)(CC / 4 / 256), 256, 0, stream>>>(
      (const float4*)Wq, (uint2*)wqh, (int)(CC / 4));
  f32_to_bf16_kernel<<<(int)(CC / 4 / 256), 256, 0, stream>>>(
      (const float4*)Wk, (uint2*)wkh, (int)(CC / 4));
  f32_to_bf16_kernel<<<(int)(CC / 4 / 256), 256, 0, stream>>>(
      (const float4*)Wv, (uint2*)wvh, (int)(CC / 4));
  f32_to_bf16_kernel<<<(int)(CC / 4 / 256), 256, 0, stream>>>(
      (const float4*)Wp, (uint2*)wph, (int)(CC / 4));

  // 2) QKV projections (Q pre-scaled by D^-1/2 = 0.125)
  dim3 gg(BN / 128, C_ / 128);
  wmma_gemm_kernel<true, false><<<gg, 256, 0, stream>>>(xh, wqh, nullptr, qh, 0.125f);
  wmma_gemm_kernel<true, false><<<gg, 256, 0, stream>>>(xh, wkh, nullptr, kh, 1.0f);
  wmma_gemm_kernel<true, false><<<gg, 256, 0, stream>>>(xh, wvh, nullptr, vh, 1.0f);

  // 3) masked scores via two grouped GEMM passes
  dim3 ga(96, H_, B_ * 2);
  attn_score_kernel<<<ga, 32, 0, stream>>>(qh, kh, S);

  // 4) softmax over the 64 allowed keys
  softmax64_kernel<<<(int)(NQS / 256), 256, 0, stream>>>(S, P, (int)NQS);

  // 5) P @ V for both key groups
  attn_pv_kernel<<<ga, 32, 0, stream>>>(P, vh, Oa, Ob);

  // 6) combine group outputs, convert to bf16
  add_to_bf16_kernel<<<(int)(BNC / 4 / 256), 256, 0, stream>>>(
      (const float4*)Oa, (const float4*)Ob, (uint2*)Oh, (int)(BNC / 4));

  // 7) output projection with bias -> fp32 d_out
  wmma_gemm_kernel<false, true><<<gg, 256, 0, stream>>>(Oh, wph, bp, out, 1.0f);
}
